// DGCNN_21242908246074
// MI455X (gfx1250) — matrix operand order repaired and checked by reading, not compiled, not run on hardware
//
#include <hip/hip_runtime.h>

// ---------------------------------------------------------------------------
// Types for WMMA fragments (gfx1250, wave32)
// ---------------------------------------------------------------------------
typedef _Float16 h8   __attribute__((ext_vector_type(8)));
typedef _Float16 v16h __attribute__((ext_vector_type(16)));
typedef float    v8f  __attribute__((ext_vector_type(8)));

union V16 { v16h v; h8 h[2]; };

#define NEG_SLOPE 0.2f
#define BN_EPS    1e-5f
#define NPTS      1024
#define KNN_K     20
#define NINF     (-3.402823466e38f)

__device__ __forceinline__ float lrelu(float x) { return x >= 0.f ? x : NEG_SLOPE * x; }

// ---------------------------------------------------------------------------
// Fold BN scale into weights, convert to f16 with zero-padded K dim.
// Wf[o][c] = (c < Cin ? W[o][c] * gamma[o]/sqrt(var[o]+eps) : 0)
// cb[o]    = beta[o] - mean[o]*s
// ---------------------------------------------------------------------------
__global__ void fold_kernel(const float* __restrict__ W,
                            const float* __restrict__ gamma,
                            const float* __restrict__ beta,
                            const float* __restrict__ mean,
                            const float* __restrict__ var,
                            int Cin, int Cp, _Float16* __restrict__ Wf,
                            float* __restrict__ cb) {
    int o = blockIdx.x;
    float s = gamma[o] * rsqrtf(var[o] + BN_EPS);
    for (int c = threadIdx.x; c < Cp; c += blockDim.x) {
        float v = (c < Cin) ? W[(size_t)o * Cin + c] * s : 0.f;
        Wf[(size_t)o * Cp + c] = (_Float16)v;
    }
    if (threadIdx.x == 0) cb[o] = beta[o] - mean[o] * s;
}

// ---------------------------------------------------------------------------
// Fused kNN + top-K selection. One wave per query point, 4 waves per block.
// pd = 2*dot(xi,xj) - |xi|^2 - |xj|^2 ; keep K=20 largest (nearest).
// Per-lane register-resident sorted top-20, then 20-round wave argmax merge.
// Reads the fp32 copy for full selection precision.
// ---------------------------------------------------------------------------
__global__ __launch_bounds__(128) void knn_kernel(const float* __restrict__ xin,
                                                  int stride, int C,
                                                  int* __restrict__ idxOut) {
    __shared__ float qs[4][132];
    const int b    = blockIdx.x >> 8;           // N/4 = 256 groups per batch
    const int n0   = (blockIdx.x & 255) * 4;
    const int wv   = threadIdx.x >> 5;
    const int lane = threadIdx.x & 31;
    const int n    = n0 + wv;

    const float* qp = xin + ((size_t)b * NPTS + n) * stride;
    for (int c = lane; c < C; c += 32) qs[wv][c] = qp[c];
    __syncthreads();

    float qn = 0.f;
    for (int c = 0; c < C; ++c) qn += qs[wv][c] * qs[wv][c];

    float val[KNN_K];
    int   id[KNN_K];
#pragma unroll
    for (int j = 0; j < KNN_K; ++j) { val[j] = NINF; id[j] = 0x7fffffff; }

    for (int m = lane; m < NPTS; m += 32) {
        const float* xj = xin + ((size_t)b * NPTS + m) * stride;
        float dot = 0.f, jn = 0.f;
        for (int c = 0; c < C; ++c) {
            float f = xj[c];
            dot += f * qs[wv][c];
            jn  += f * f;
        }
        float pd = 2.f * dot - qn - jn;
        if (pd > val[KNN_K - 1]) {
            val[KNN_K - 1] = pd; id[KNN_K - 1] = m;
#pragma unroll
            for (int j = KNN_K - 1; j > 0; --j) {
                if (val[j] > val[j - 1]) {
                    float tv = val[j]; val[j] = val[j - 1]; val[j - 1] = tv;
                    int   ti = id[j];  id[j]  = id[j - 1];  id[j - 1]  = ti;
                }
            }
        }
    }

    // merge 32 sorted lists -> global top-20 (descending, tie -> smaller index)
    int* out = idxOut + ((size_t)b * NPTS + n) * KNN_K;
    for (int r = 0; r < KNN_K; ++r) {
        float v = val[0]; int ii = id[0];
#pragma unroll
        for (int off = 16; off > 0; off >>= 1) {
            float ov = __shfl_xor(v, off, 32);
            int   oi = __shfl_xor(ii, off, 32);
            if (ov > v || (ov == v && oi < ii)) { v = ov; ii = oi; }
        }
        if (lane == 0) out[r] = ii;
        if (id[0] == ii) {   // winner pops its head (indices unique across lanes)
#pragma unroll
            for (int j = 0; j < KNN_K - 1; ++j) { val[j] = val[j + 1]; id[j] = id[j + 1]; }
            val[KNN_K - 1] = NINF; id[KNN_K - 1] = 0x7fffffff;
        }
    }
}

// ---------------------------------------------------------------------------
// EdgeConv: gather cat(nbr-ctr, ctr) into LDS f16, WMMA GEMM against folded
// f16 weights, bias + LReLU in-register, max over K=20 via LDS.
// Block: 160 threads = 5 waves; handles 4 points = 80 feat rows = 5 M-tiles.
//   KC = (2C padded to 32)/32 k-chunks, OT = O/16 output tiles.
//   USEH: gather from the f16 feature copy (layers 2-4); layer 1 uses fp32 x.
// Writes both the fp32 copy (for kNN) and f16 copy (for gather / W5 GEMM).
// ---------------------------------------------------------------------------
template <int KC, int OT, bool USEH>
__global__ __launch_bounds__(160) void edge_kernel(const float* __restrict__ xinF,
                                                   const _Float16* __restrict__ xinH,
                                                   int stride, int C,
                                                   const int* __restrict__ idxAll,
                                                   const _Float16* __restrict__ Wf,
                                                   const float* __restrict__ cb,
                                                   float* __restrict__ outF,
                                                   _Float16* __restrict__ outH, int Ooff) {
    constexpr int Cp2 = KC * 32;
    __shared__ __align__(16) _Float16 feat[80 * Cp2];
    __shared__ float hbuf[80 * 16];
    __shared__ int   sidx[80];

    const int b   = blockIdx.x >> 8;
    const int n0  = (blockIdx.x & 255) * 4;
    const int tid = threadIdx.x;

    for (int e = tid; e < 80; e += 160) {
        int n = n0 + e / KNN_K;
        sidx[e] = idxAll[((size_t)b * NPTS + n) * KNN_K + (e % KNN_K)];
    }
    __syncthreads();

    // Build feature tile: rows r = (n,k) flattened, cols = [nbr-ctr | ctr | 0pad]
    for (int e = tid; e < 80 * Cp2; e += 160) {
        int r = e / Cp2, c = e % Cp2;
        int n = n0 + r / KNN_K;
        int j = sidx[r];
        _Float16 hv;
        if (USEH) {
            if (c < C)
                hv = (_Float16)(xinH[((size_t)b * NPTS + j) * stride + c] -
                                xinH[((size_t)b * NPTS + n) * stride + c]);
            else if (c < 2 * C)
                hv = xinH[((size_t)b * NPTS + n) * stride + (c - C)];
            else
                hv = (_Float16)0.f;
        } else {
            float v;
            if (c < C)
                v = xinF[((size_t)b * NPTS + j) * stride + c] -
                    xinF[((size_t)b * NPTS + n) * stride + c];
            else if (c < 2 * C)
                v = xinF[((size_t)b * NPTS + n) * stride + (c - C)];
            else
                v = 0.f;
            hv = (_Float16)v;
        }
        feat[r * Cp2 + c] = hv;
    }
    __syncthreads();

    const int wv   = tid >> 5;      // wave owns M-tile wv (rows 16wv..16wv+15)
    const int lane = tid & 31;
    const int mrow = wv * 16 + (lane & 15);
    const int hh   = lane >> 4;     // lane half

    // Hoist A fragments (ISA 16-bit A layout: runs [koff,+8) and [koff+16,+8))
    v16h af[KC];
#pragma unroll
    for (int kc = 0; kc < KC; ++kc) {
        int koff = kc * 32 + 8 * hh;
        V16 a;
        a.h[0] = *(const h8*)&feat[mrow * Cp2 + koff];
        a.h[1] = *(const h8*)&feat[mrow * Cp2 + koff + 16];
        af[kc] = a.v;
    }

    for (int ot = 0; ot < OT; ++ot) {
        v8f acc = {0.f, 0.f, 0.f, 0.f, 0.f, 0.f, 0.f, 0.f};
        const int o = ot * 16 + (lane & 15);
#pragma unroll
        for (int kc = 0; kc < KC; ++kc) {
            // B layout: same as A, runs [kc*32+8hh,+8) and [kc*32+8hh+16,+8)
            const _Float16* wp = Wf + (size_t)o * Cp2 + kc * 32 + 8 * hh;
            V16 bf;
            bf.h[0] = *(const h8*)wp;
            bf.h[1] = *(const h8*)(wp + 16);
            acc = __builtin_amdgcn_wmma_f32_16x16x32_f16(
                false, af[kc], false, bf.v, (short)0, acc, false, false);
        }
        const float cbo = cb[o];
#pragma unroll
        for (int v = 0; v < 8; ++v) {
            float x = lrelu(acc[v] + cbo);
            hbuf[(wv * 16 + v + 8 * hh) * 16 + (lane & 15)] = x;  // D layout
        }
        __syncthreads();
        if (tid < 64) {  // max over K neighbors, write cat slice (fp32 + f16)
            int g = tid >> 4, oc = tid & 15;
            float mx = NINF;
#pragma unroll
            for (int kk = 0; kk < KNN_K; ++kk)
                mx = fmaxf(mx, hbuf[(g * KNN_K + kk) * 16 + oc]);
            size_t oidx = ((size_t)b * NPTS + n0 + g) * 512 + Ooff + ot * 16 + oc;
            outF[oidx] = mx;
            outH[oidx] = (_Float16)mx;
        }
        __syncthreads();
    }
}

// ---------------------------------------------------------------------------
// W5 projection (512 -> 1024) fused with global max+sum pooling over N.
// Block: 128 threads = 4 waves; wave owns one 16-wide output strip for one b.
// B fragments (16 k-chunks) hoisted in registers; A streams the f16 cat copy
// (pure global_load_b128, no conversions); prefetch hints for upcoming rows.
// ---------------------------------------------------------------------------
__global__ __launch_bounds__(128) void w5_pool_kernel(const _Float16* __restrict__ catH,
                                                      const _Float16* __restrict__ W5f,
                                                      const float* __restrict__ c5,
                                                      float* __restrict__ gmax,
                                                      float* __restrict__ gsum) {
    const int b      = blockIdx.x >> 4;
    const int og     = blockIdx.x & 15;
    const int wv     = threadIdx.x >> 5;
    const int lane   = threadIdx.x & 31;
    const int hh     = lane >> 4;
    const int ostrip = og * 4 + wv;              // 0..63
    const int o      = ostrip * 16 + (lane & 15);

    v16h bf[16];
#pragma unroll
    for (int kc = 0; kc < 16; ++kc) {
        const _Float16* wp = W5f + (size_t)o * 512 + kc * 32 + 8 * hh;
        V16 t;
        t.h[0] = *(const h8*)wp;
        t.h[1] = *(const h8*)(wp + 16);
        bf[kc] = t.v;
    }

    float rmax[8], rsum[8];
#pragma unroll
    for (int v = 0; v < 8; ++v) { rmax[v] = NINF; rsum[v] = 0.f; }
    const float cbo = c5[o];

    for (int nt = 0; nt < 64; ++nt) {
        const int row = nt * 16 + (lane & 15);
        // Latency hint for the row two tiles ahead (lowered to global_prefetch_b8)
        if (nt + 2 < 64)
            __builtin_prefetch(catH + ((size_t)b * NPTS + row + 32) * 512, 0, 0);
        v8f acc = {0.f, 0.f, 0.f, 0.f, 0.f, 0.f, 0.f, 0.f};
        const _Float16* rp = catH + ((size_t)b * NPTS + row) * 512 + 8 * hh;
#pragma unroll
        for (int kc = 0; kc < 16; ++kc) {
            const _Float16* p = rp + kc * 32;
            V16 a;
            a.h[0] = *(const h8*)p;          // K = koff .. koff+7
            a.h[1] = *(const h8*)(p + 16);   // K = koff+16 .. koff+23
            acc = __builtin_amdgcn_wmma_f32_16x16x32_f16(
                false, a.v, false, bf[kc], (short)0, acc, false, false);
        }
#pragma unroll
        for (int v = 0; v < 8; ++v) {
            float x = lrelu(acc[v] + cbo);
            rmax[v] = fmaxf(rmax[v], x);
            rsum[v] += x;
        }
    }
    float m = NINF, s = 0.f;
#pragma unroll
    for (int v = 0; v < 8; ++v) { m = fmaxf(m, rmax[v]); s += rsum[v]; }
    m = fmaxf(m, __shfl_xor(m, 16, 32));   // combine row halves (same column o)
    s += __shfl_xor(s, 16, 32);
    if (lane < 16) {
        gmax[(size_t)b * 1024 + o] = m;
        gsum[(size_t)b * 1024 + o] = s;
    }
}

// ---------------------------------------------------------------------------
// Head MLP (tiny): scalar fp32.
// ---------------------------------------------------------------------------
__global__ void head1_kernel(const float* __restrict__ gmax, const float* __restrict__ gsum,
                             const float* __restrict__ lin1,
                             const float* __restrict__ g6, const float* __restrict__ b6,
                             const float* __restrict__ m6, const float* __restrict__ v6,
                             float* __restrict__ h1) {
    int r = blockIdx.x, o = threadIdx.x;  // 512 threads
    const float* wrow = lin1 + (size_t)o * 2048;
    const float* gm = gmax + (size_t)r * 1024;
    const float* gs = gsum + (size_t)r * 1024;
    float acc = 0.f;
    for (int k = 0; k < 1024; ++k) acc += gm[k] * wrow[k];
    const float inv = 1.0f / (float)NPTS;
    for (int k = 0; k < 1024; ++k) acc += gs[k] * inv * wrow[1024 + k];
    float s = g6[o] * rsqrtf(v6[o] + BN_EPS);
    h1[(size_t)r * 512 + o] = lrelu((acc - m6[o]) * s + b6[o]);
}

__global__ void head2_kernel(const float* __restrict__ h1, const float* __restrict__ lin2,
                             const float* __restrict__ b2,
                             const float* __restrict__ g7, const float* __restrict__ b7,
                             const float* __restrict__ m7, const float* __restrict__ v7,
                             float* __restrict__ h2) {
    int r = blockIdx.x, o = threadIdx.x;  // 256 threads
    const float* wrow = lin2 + (size_t)o * 512;
    const float* hr = h1 + (size_t)r * 512;
    float acc = b2[o];
    for (int k = 0; k < 512; ++k) acc += hr[k] * wrow[k];
    float s = g7[o] * rsqrtf(v7[o] + BN_EPS);
    h2[(size_t)r * 256 + o] = lrelu((acc - m7[o]) * s + b7[o]);
}

__global__ void head3_kernel(const float* __restrict__ h2, const float* __restrict__ lin3,
                             const float* __restrict__ b3, float* __restrict__ outp) {
    int r = blockIdx.x, o = threadIdx.x;  // 64 threads, guard 40
    if (o >= 40) return;
    const float* wrow = lin3 + (size_t)o * 256;
    const float* hr = h2 + (size_t)r * 256;
    float acc = b3[o];
    for (int k = 0; k < 256; ++k) acc += hr[k] * wrow[k];
    outp[(size_t)r * 40 + o] = acc;
}

// ---------------------------------------------------------------------------
// Host launcher
// ---------------------------------------------------------------------------
static inline size_t align256(size_t x) { return (x + 255) & ~(size_t)255; }

extern "C" void kernel_launch(void* const* d_in, const int* in_sizes, int n_in,
                              void* d_out, int out_size, void* d_ws, size_t ws_size,
                              hipStream_t stream) {
    (void)in_sizes; (void)n_in; (void)out_size; (void)ws_size;

    // setup_inputs() flatten order: x, W1, bn1{g,b,m,v}, W2, bn2, W3, bn3,
    // W4, bn4, W5, bn5, lin1, bn6, lin2, b2, bn7, lin3, b3
    const float* X  = (const float*)d_in[0];
    const float* W1 = (const float*)d_in[1];
    const float *g1 = (const float*)d_in[2],  *b1 = (const float*)d_in[3],
                *m1 = (const float*)d_in[4],  *v1 = (const float*)d_in[5];
    const float* W2 = (const float*)d_in[6];
    const float *g2 = (const float*)d_in[7],  *b2_ = (const float*)d_in[8],
                *m2 = (const float*)d_in[9],  *v2 = (const float*)d_in[10];
    const float* W3 = (const float*)d_in[11];
    const float *g3 = (const float*)d_in[12], *b3_ = (const float*)d_in[13],
                *m3 = (const float*)d_in[14], *v3 = (const float*)d_in[15];
    const float* W4 = (const float*)d_in[16];
    const float *g4 = (const float*)d_in[17], *b4_ = (const float*)d_in[18],
                *m4 = (const float*)d_in[19], *v4 = (const float*)d_in[20];
    const float* W5 = (const float*)d_in[21];
    const float *g5 = (const float*)d_in[22], *b5_ = (const float*)d_in[23],
                *m5 = (const float*)d_in[24], *v5 = (const float*)d_in[25];
    const float* LIN1 = (const float*)d_in[26];
    const float *g6 = (const float*)d_in[27], *b6 = (const float*)d_in[28],
                *m6 = (const float*)d_in[29], *v6 = (const float*)d_in[30];
    const float* LIN2 = (const float*)d_in[31];
    const float* B2   = (const float*)d_in[32];
    const float *g7 = (const float*)d_in[33], *b7 = (const float*)d_in[34],
                *m7 = (const float*)d_in[35], *v7 = (const float*)d_in[36];
    const float* LIN3 = (const float*)d_in[37];
    const float* B3   = (const float*)d_in[38];

    // Workspace carve-up
    char* ws = (char*)d_ws;
    size_t off = 0;
    float* cat = (float*)(ws + off);          off = align256(off + (size_t)32 * 1024 * 512 * 4);
    _Float16* catH = (_Float16*)(ws + off);   off = align256(off + (size_t)32 * 1024 * 512 * 2);
    int* idx   = (int*)(ws + off);            off = align256(off + (size_t)32 * 1024 * KNN_K * 4);
    _Float16* W1f = (_Float16*)(ws + off);    off = align256(off + (size_t)64 * 32 * 2);
    _Float16* W2f = (_Float16*)(ws + off);    off = align256(off + (size_t)64 * 128 * 2);
    _Float16* W3f = (_Float16*)(ws + off);    off = align256(off + (size_t)128 * 128 * 2);
    _Float16* W4f = (_Float16*)(ws + off);    off = align256(off + (size_t)256 * 256 * 2);
    _Float16* W5f = (_Float16*)(ws + off);    off = align256(off + (size_t)1024 * 512 * 2);
    float* c1 = (float*)(ws + off);           off = align256(off + 64 * 4);
    float* c2 = (float*)(ws + off);           off = align256(off + 64 * 4);
    float* c3 = (float*)(ws + off);           off = align256(off + 128 * 4);
    float* c4 = (float*)(ws + off);           off = align256(off + 256 * 4);
    float* c5 = (float*)(ws + off);           off = align256(off + 1024 * 4);
    float* gmax = (float*)(ws + off);         off = align256(off + (size_t)32 * 1024 * 4);
    float* gsum = (float*)(ws + off);         off = align256(off + (size_t)32 * 1024 * 4);
    float* h1 = (float*)(ws + off);           off = align256(off + (size_t)32 * 512 * 4);
    float* h2 = (float*)(ws + off);           off = align256(off + (size_t)32 * 256 * 4);

    // Fold BN into f16 weights
    fold_kernel<<<64,   128, 0, stream>>>(W1, g1, b1, m1, v1,   6,  32, W1f, c1);
    fold_kernel<<<64,   128, 0, stream>>>(W2, g2, b2_, m2, v2, 128, 128, W2f, c2);
    fold_kernel<<<128,  128, 0, stream>>>(W3, g3, b3_, m3, v3, 128, 128, W3f, c3);
    fold_kernel<<<256,  128, 0, stream>>>(W4, g4, b4_, m4, v4, 256, 256, W4f, c4);
    fold_kernel<<<1024, 128, 0, stream>>>(W5, g5, b5_, m5, v5, 512, 512, W5f, c5);

    const int nGrid = 32 * 256;  // B * N/4

    // EdgeConv stack; outputs written into cat slices [0|64|128|256..512),
    // in both fp32 (kNN precision) and f16 (gather + W5 GEMM operand) copies.
    knn_kernel<<<nGrid, 128, 0, stream>>>(X, 3, 3, idx);
    edge_kernel<1, 4, false><<<nGrid, 160, 0, stream>>>(
        X, (const _Float16*)nullptr, 3, 3, idx, W1f, c1, cat, catH, 0);

    knn_kernel<<<nGrid, 128, 0, stream>>>(cat, 512, 64, idx);
    edge_kernel<4, 4, true><<<nGrid, 160, 0, stream>>>(
        cat, catH, 512, 64, idx, W2f, c2, cat, catH, 64);

    knn_kernel<<<nGrid, 128, 0, stream>>>(cat + 64, 512, 64, idx);
    edge_kernel<4, 8, true><<<nGrid, 160, 0, stream>>>(
        cat + 64, catH + 64, 512, 64, idx, W3f, c3, cat, catH, 128);

    knn_kernel<<<nGrid, 128, 0, stream>>>(cat + 128, 512, 128, idx);
    edge_kernel<8, 16, true><<<nGrid, 160, 0, stream>>>(
        cat + 128, catH + 128, 512, 128, idx, W4f, c4, cat, catH, 256);

    // W5 + fused global max/mean pooling (streams the f16 cat copy)
    w5_pool_kernel<<<32 * 16, 128, 0, stream>>>(catH, W5f, c5, gmax, gsum);

    // Head MLP
    head1_kernel<<<32, 512, 0, stream>>>(gmax, gsum, LIN1, g6, b6, m6, v6, h1);
    head2_kernel<<<32, 256, 0, stream>>>(h1, LIN2, B2, g7, b7, m7, v7, h2);
    head3_kernel<<<32, 64, 0, stream>>>(h2, LIN3, B3, (float*)d_out);
}
